// Generator_67748814127441
// MI455X (gfx1250) — matrix operand order repaired and checked by reading, compiled, not run on hardware
//
#include <hip/hip_runtime.h>
#include <hip/hip_bf16.h>
#include <math.h>

// ---------------------------------------------------------------------------
// Problem constants (from reference): B=8, CIN=COUT=512, K=3, G=16, SD=512,
// H=32 -> output [8,512,64,64].  Fused transposed-conv+blur via parity
// decomposition -> per (sample,parity) GEMM  M=512, N=1024, K=512*9=4608.
// ---------------------------------------------------------------------------

typedef __attribute__((ext_vector_type(16))) __bf16 v16bf;
typedef __attribute__((ext_vector_type(8)))  float  v8f;

__device__ __forceinline__ float sigmoidf(float x) { return 1.0f / (1.0f + __expf(-x)); }

// ---------------- workspace layout (bytes) ----------------
#define OFF_CL     ((size_t)0)           // cl[16][512] f32            32 KB
#define OFF_STS    ((size_t)32768)       // stscale[8][512] f32        16 KB
#define OFF_STV    ((size_t)49152)       // stvec[8][512] f32          16 KB
#define OFF_DEMOD  ((size_t)65536)       // demod[8][512] f32          16 KB
#define OFF_RN     ((size_t)81920)       // rn[512] f32                 2 KB
#define OFF_ACC    ((size_t)83968)       // acc[2] f32 (loss partials)
#define OFF_WSQ    ((size_t)1 << 20)     // Wsq[512][512] f32           1 MB
#define OFF_WT     ((size_t)2 << 20)     // WT[4][9][512][512] bf16  18.9 MB
#define OFF_BB     ((size_t)24 << 20)    // Bb[8][9][1024][512] bf16 75.5 MB

// ===========================================================================
// P0: style pipeline (single block, 512 threads; thread i = channel i)
// ===========================================================================
__global__ __launch_bounds__(512) void k_style(
    const float* __restrict__ style, const float* __restrict__ mb_w,
    const float* __restrict__ mb_b,  const float* __restrict__ mod_w,
    const float* __restrict__ mod_b, const float* __restrict__ cluster,
    const float* __restrict__ tmpr,
    float* __restrict__ cl_ws, float* __restrict__ stscale,
    float* __restrict__ stvec, float* __restrict__ acc)
{
    __shared__ float s_grp2[16][512];
    __shared__ float s_st02[8][512];
    __shared__ float s_coef[8][16];
    const int i = threadIdx.x;
    const float lin = 1.0f / sqrtf(512.0f);
    const float t = sigmoidf(tmpr[0]) * 5.0f;

    // cl[:,i] = softmax_g(t*cluster[:,i]) ; grouped = round(softmax(cl*500))+1e-6
    float c[16], mx = -1e30f;
    #pragma unroll
    for (int g = 0; g < 16; ++g) { c[g] = t * cluster[g * 512 + i]; mx = fmaxf(mx, c[g]); }
    float se = 0.f;
    #pragma unroll
    for (int g = 0; g < 16; ++g) { c[g] = __expf(c[g] - mx); se += c[g]; }
    const float inv = 1.f / se;
    float grp[16], mx2 = -1e30f;
    #pragma unroll
    for (int g = 0; g < 16; ++g) { c[g] *= inv; cl_ws[g * 512 + i] = c[g]; mx2 = fmaxf(mx2, c[g]); }
    float se2 = 0.f;
    #pragma unroll
    for (int g = 0; g < 16; ++g) { grp[g] = __expf(500.f * (c[g] - mx2)); se2 += grp[g]; }
    #pragma unroll
    for (int g = 0; g < 16; ++g) { grp[g] = roundf(grp[g] / se2) + 1e-6f; s_grp2[g][i] = grp[g] * grp[g]; }

    // st0[b,i] = s2[b] . mod_w[i,:] * lin + mod_b[i]
    float st0[8];
    #pragma unroll
    for (int b = 0; b < 8; ++b) {
        const float* s2 = style + b * 1024 + 512;
        float s = 0.f;
        for (int k = 0; k < 512; ++k) s += s2[k] * mod_w[i * 512 + k];
        st0[b] = s * lin + mod_b[i];
        s_st02[b][i] = st0[b] * st0[b];
    }
    __syncthreads();

    // coef[b,g] = style_base[b,g] / ||grouped[g,:]*st0[b,:]||
    if (i < 128) {
        const int b = i >> 4, gq = i & 15;
        float ss = 0.f;
        for (int k = 0; k < 512; ++k) ss += s_grp2[gq][k] * s_st02[b][k];
        const float den = fmaxf(sqrtf(ss), 1e-12f);
        const float* s1 = style + b * 1024;
        float sb = 0.f;
        for (int k = 0; k < 512; ++k) sb += s1[k] * mb_w[gq * 512 + k];
        sb = sb * lin + mb_b[gq];
        s_coef[b][gq] = sb / den;
    }
    if (i < 2) acc[i] = 0.f;   // zero loss accumulators every launch
    __syncthreads();

    const float scale = 1.0f / sqrtf(4608.0f);   // 1/sqrt(CIN*K*K)
    #pragma unroll
    for (int b = 0; b < 8; ++b) {
        float m = 0.f;
        #pragma unroll
        for (int g = 0; g < 16; ++g) m += grp[g] * s_coef[b][g];
        const float sv = st0[b] * m;
        stvec[b * 512 + i] = sv;
        stscale[b * 512 + i] = sv * scale;   // folded into activations
    }
}

// ===========================================================================
// P1: Wsq[o,i] = sum_t W^2 ; WT[p,tap,o,i] = (T[p]W)[tap]  (bf16 GEMM A)
// 1-D blur*upsample tap tables t1[r][d][j] (over k=[1,3,3,1], /4 per axis)
// ===========================================================================
__global__ __launch_bounds__(256) void k_wsq_wt(
    const float* __restrict__ W, float* __restrict__ Wsq, __bf16* __restrict__ WT)
{
    const int idx = blockIdx.x * 256 + threadIdx.x;     // (o,i) pair, 512*512
    const int o = idx >> 9, i = idx & 511;
    float w[9], sq = 0.f;
    #pragma unroll
    for (int tt = 0; tt < 9; ++tt) { w[tt] = W[(size_t)idx * 9 + tt]; sq += w[tt] * w[tt]; }
    Wsq[idx] = sq;

    const float t1[2][3][3] = { { {0,1,3}, {3,3,1}, {1,0,0} },
                                { {0,0,1}, {1,3,3}, {3,1,0} } };
    #pragma unroll
    for (int ry = 0; ry < 2; ++ry)
    #pragma unroll
    for (int rx = 0; rx < 2; ++rx) {
        const int p = ry * 2 + rx;
        #pragma unroll
        for (int dy = 0; dy < 3; ++dy)
        #pragma unroll
        for (int dx = 0; dx < 3; ++dx) {
            float s = 0.f;
            #pragma unroll
            for (int jy = 0; jy < 3; ++jy)
            #pragma unroll
            for (int jx = 0; jx < 3; ++jx)
                s += t1[ry][dy][jy] * t1[rx][dx][jx] * w[jy * 3 + jx];
            s *= (1.0f / 16.0f);
            const int tap = dy * 3 + dx;
            WT[((size_t)(p * 9 + tap) * 512 + o) * 512 + i] = (__bf16)s;
        }
    }
}

// rn[o] = 1/max(||W[o,:]||, 1e-12) from Wsq row sums
__global__ void k_rn(const float* __restrict__ Wsq, float* __restrict__ rn)
{
    const int o = blockIdx.x * blockDim.x + threadIdx.x;
    if (o >= 512) return;
    float s = 0.f;
    for (int i = 0; i < 512; ++i) s += Wsq[o * 512 + i];
    rn[o] = 1.0f / fmaxf(sqrtf(s), 1e-12f);
}

// demod[b,o] = rsqrt(scale^2 * sum_i st^2 * Wsq + 1e-8)
__global__ void k_demod(const float* __restrict__ Wsq, const float* __restrict__ stvec,
                        float* __restrict__ demod)
{
    const int t = blockIdx.x * blockDim.x + threadIdx.x;   // 4096
    const int b = t >> 9, o = t & 511;
    float s = 0.f;
    for (int i = 0; i < 512; ++i) {
        const float sv = stvec[b * 512 + i];
        s += sv * sv * Wsq[o * 512 + i];
    }
    demod[t] = 1.0f / sqrtf(s * (1.0f / 4608.0f) + 1e-8f);
}

// ===========================================================================
// P3: im2col (tap-major) with style-scale folded in, bf16, channel-contiguous
// Bb[b][tap][n=sy*32+sx][i] = x[b,i,sy+dy-1,sx+dx-1] * scale*st[b,i]  (0 pad)
// ===========================================================================
__global__ __launch_bounds__(512) void k_bbuf(
    const float* __restrict__ x, const float* __restrict__ stscale, __bf16* __restrict__ Bb)
{
    const int i = threadIdx.x;
    const int n = blockIdx.x, tap = blockIdx.y, b = blockIdx.z;
    const int sy = n >> 5, sx = n & 31;
    const int yy = sy + tap / 3 - 1, xx = sx + tap % 3 - 1;
    float v = 0.f;
    if ((unsigned)yy < 32u && (unsigned)xx < 32u)
        v = x[((size_t)(b * 512 + i) * 32 + yy) * 32 + xx] * stscale[b * 512 + i];
    Bb[((size_t)(b * 9 + tap) * 1024 + n) * 512 + i] = (__bf16)v;
}

// ===========================================================================
// Losses: loss1 = sum((cl^T cl - S)^2), loss2 = (16384 - sum S)^2
// ===========================================================================
__global__ __launch_bounds__(256) void k_loss(
    const float* __restrict__ co, const float* __restrict__ cl,
    const float* __restrict__ gate, float* __restrict__ acc)
{
    const int tid = threadIdx.x;
    const int idx = blockIdx.x * 256 + tid;            // 262144 = 512*512
    const int i = idx >> 9, j = idx & 511;
    const float g = sigmoidf(gate[0]);
    const float S = sigmoidf((co[idx] - g) * 50.0f);
    float d = 0.f;
    #pragma unroll
    for (int q = 0; q < 16; ++q) d += cl[q * 512 + i] * cl[q * 512 + j];
    const float e = d - S;
    __shared__ float r1[256], r2[256];
    r1[tid] = e * e; r2[tid] = S;
    __syncthreads();
    for (int s = 128; s > 0; s >>= 1) {
        if (tid < s) { r1[tid] += r1[tid + s]; r2[tid] += r2[tid + s]; }
        __syncthreads();
    }
    if (tid == 0) { atomicAdd(&acc[0], r1[0]); atomicAdd(&acc[1], r2[0]); }
}

__global__ void k_final(const float* __restrict__ acc, float* __restrict__ loss_out)
{
    loss_out[0] = acc[0];
    const float d = 16384.0f - acc[1];
    loss_out[1] = d * d;
}

// ===========================================================================
// Gram: co_out = l2norm(W.reshape(512,4608)) @ ^T   (1.2 GMAC, f32 VALU)
// ===========================================================================
__global__ __launch_bounds__(256) void k_gram(
    const float* __restrict__ W, const float* __restrict__ rn, float* __restrict__ co_out)
{
    __shared__ float As[16][17], Bs[16][17];
    const int tx = threadIdx.x & 15, ty = threadIdx.x >> 4;
    const int ib = blockIdx.x * 16, jb = blockIdx.y * 16;
    float a = 0.f;
    for (int k0 = 0; k0 < 4608; k0 += 16) {
        As[ty][tx] = W[(size_t)(ib + ty) * 4608 + k0 + tx];
        Bs[ty][tx] = W[(size_t)(jb + ty) * 4608 + k0 + tx];
        __syncthreads();
        #pragma unroll
        for (int k = 0; k < 16; ++k) a += As[ty][k] * Bs[tx][k];
        __syncthreads();
    }
    co_out[(ib + ty) * 512 + jb + tx] = a * rn[ib + ty] * rn[jb + tx];
}

// ===========================================================================
// Main fused conv as implicit GEMM with v_wmma_f32_16x16x32_bf16 and
// GLOBAL_LOAD_ASYNC_TO_LDS_B128 double-buffered staging (ASYNCcnt).
// grid = (Mtiles=4, Ntiles=8, b*4+parity=32); block = 256 (8 waves, wave32).
// Tile: 128x128; wave tile 32x64 = 2x4 accum frags; K-loop 144 chunks of 32.
// ===========================================================================
#define LDA   40                 // padded LDS row stride (bf16) to spread banks
#define BUFE  (128 * LDA)        // elements per LDS buffer (one matrix)

__device__ __forceinline__ void async_ld_2x16(const __bf16* g, unsigned lds_addr)
{
    // 32 contiguous bytes: two b128 async copies; INST_OFFSET applies to both
    // the LDS and global side (ISA 08_async_tensor §4.4), so offset:16 moves both.
    asm volatile("global_load_async_to_lds_b128 %0, %1, off"
                 :: "v"(lds_addr), "v"(g) : "memory");
    asm volatile("global_load_async_to_lds_b128 %0, %1, off offset:16"
                 :: "v"(lds_addr), "v"(g) : "memory");
}
__device__ __forceinline__ void wait_async0()
{
    asm volatile("s_wait_asynccnt 0x0" ::: "memory");
}

__global__ __launch_bounds__(256) void k_conv(
    const __bf16* __restrict__ WT, const __bf16* __restrict__ Bb,
    const float* __restrict__ demod, const float* __restrict__ noise,
    const float* __restrict__ flr_bias, const float* __restrict__ noise_w,
    float* __restrict__ out)
{
    __shared__ __bf16 lA[2 * BUFE];
    __shared__ __bf16 lB[2 * BUFE];
    const int tid = threadIdx.x;
    const int lane = tid & 31, wave = tid >> 5;
    const int wm = wave & 3, wn = wave >> 2;           // 4 x 2 wave grid
    const int bz = blockIdx.z;
    const int b = bz >> 2, par = bz & 3;
    const int m0 = blockIdx.x * 128;                   // output-channel tile
    const int n0 = blockIdx.y * 128;                   // spatial tile

    // staging: thread -> (row, 16-elem half); 128 rows * 2 halves = 256 slots
    const int sr = tid >> 1;
    const int sh = (tid & 1) * 16;
    const __bf16* Abase = WT + (size_t)par * 9 * 512 * 512 + (size_t)(m0 + sr) * 512 + sh;
    const __bf16* Bbase = Bb + (size_t)b * 9 * 1024 * 512 + (size_t)(n0 + sr) * 512 + sh;
    const unsigned ldsA = (unsigned)(size_t)&lA[sr * LDA + sh];   // low 32 bits = LDS addr
    const unsigned ldsB = (unsigned)(size_t)&lB[sr * LDA + sh];

    auto issue_chunk = [&](int kc, int buf) {
        const int tap = kc >> 4;
        const int i0 = (kc & 15) * 32;
        async_ld_2x16(Abase + (size_t)tap * 512 * 512  + i0, ldsA + buf * (BUFE * 2));
        async_ld_2x16(Bbase + (size_t)tap * 1024 * 512 + i0, ldsB + buf * (BUFE * 2));
    };

    v8f acc[2][4] = {};
    const int hi = lane >> 4, ln = lane & 15;

    int buf = 0;
    issue_chunk(0, buf);
    wait_async0();
    __syncthreads();

    for (int kc = 0; kc < 144; ++kc) {
        if (kc + 1 < 144) issue_chunk(kc + 1, buf ^ 1);
        if (kc + 2 < 144) {           // gfx1250 global_prefetch_b8, nearer scope
            const int t2 = (kc + 2) >> 4, i2 = ((kc + 2) & 15) * 32;
            __builtin_prefetch(Abase + (size_t)t2 * 512 * 512 + i2, 0, 3);
            __builtin_prefetch(Bbase + (size_t)t2 * 1024 * 512 + i2, 0, 3);
        }

        const __bf16* bufA = &lA[buf * BUFE];
        const __bf16* bufB = &lB[buf * BUFE];
        // A fragments: lane<16 -> row m=ln, K {0..7,16..23}; lane>=16 -> K {8..15,24..31}
        v16bf af[2], bfr[4];
        #pragma unroll
        for (int mi = 0; mi < 2; ++mi) {
            const int r = wm * 32 + mi * 16 + ln;
            const uint4* p = (const uint4*)&bufA[r * LDA];
            union { uint4 q[2]; v16bf v; } u;
            u.q[0] = p[hi]; u.q[1] = p[2 + hi];
            af[mi] = u.v;
        }
        // B fragments: lane holds column n=ln, K {0..7,16..23} (lane<16) / {8..15,24..31} (lane>=16)
        #pragma unroll
        for (int ni = 0; ni < 4; ++ni) {
            const int cidx = wn * 64 + ni * 16 + ln;
            const uint4* p = (const uint4*)&bufB[cidx * LDA];
            union { uint4 q[2]; v16bf v; } u;
            u.q[0] = p[hi]; u.q[1] = p[2 + hi];
            bfr[ni] = u.v;
        }
        #pragma unroll
        for (int mi = 0; mi < 2; ++mi)
        #pragma unroll
        for (int ni = 0; ni < 4; ++ni)
            acc[mi][ni] = __builtin_amdgcn_wmma_f32_16x16x32_bf16(
                false, af[mi], false, bfr[ni], (short)0, acc[mi][ni], false, false);

        wait_async0();       // this wave's async stores for chunk kc+1 landed in LDS
        __syncthreads();     // publish to all waves; also fences reads of old buf
        buf ^= 1;
    }

    // Epilogue: *demod, +noise, +bias, leaky_relu*sqrt(2), parity interleave
    const float nw = noise_w[0];
    const float sq2 = 1.41421356237f;
    const int ry = par >> 1, rx = par & 1;
    #pragma unroll
    for (int mi = 0; mi < 2; ++mi) {
        #pragma unroll
        for (int ni = 0; ni < 4; ++ni) {
            const int n = n0 + wn * 64 + ni * 16 + ln;
            const int sy = n >> 5, sx = n & 31;
            const int oy = 2 * sy + ry, ox = 2 * sx + rx;
            const float nz = nw * noise[(b * 64 + oy) * 64 + ox];
            #pragma unroll
            for (int v = 0; v < 8; ++v) {
                const int m = m0 + wm * 32 + mi * 16 + hi * 8 + v;
                float val = acc[mi][ni][v] * demod[b * 512 + m] + nz + flr_bias[m];
                val = (val > 0.f ? val : 0.2f * val) * sq2;
                out[((size_t)(b * 512 + m) * 64 + oy) * 64 + ox] = val;
            }
        }
    }
}

// ===========================================================================
extern "C" void kernel_launch(void* const* d_in, const int* in_sizes, int n_in,
                              void* d_out, int out_size, void* d_ws, size_t ws_size,
                              hipStream_t stream)
{
    (void)in_sizes; (void)n_in; (void)out_size; (void)ws_size;
    const float* x        = (const float*)d_in[0];
    const float* co       = (const float*)d_in[1];
    const float* style    = (const float*)d_in[2];
    const float* noise    = (const float*)d_in[3];
    const float* W        = (const float*)d_in[4];
    const float* mb_w     = (const float*)d_in[5];
    const float* mb_b     = (const float*)d_in[6];
    const float* mod_w    = (const float*)d_in[7];
    const float* mod_b    = (const float*)d_in[8];
    const float* cluster  = (const float*)d_in[9];
    const float* gate     = (const float*)d_in[10];
    const float* tmpr     = (const float*)d_in[11];
    const float* noise_w  = (const float*)d_in[12];
    const float* flr_bias = (const float*)d_in[13];

    float* out_main = (float*)d_out;                       // [8,512,64,64]
    float* co_out   = (float*)d_out + 16777216;            // [1,512,512]
    float* loss_out = (float*)d_out + 16777216 + 262144;   // loss1, loss2

    char* ws = (char*)d_ws;
    float*  cl_ws   = (float*)(ws + OFF_CL);
    float*  stscale = (float*)(ws + OFF_STS);
    float*  stvec   = (float*)(ws + OFF_STV);
    float*  demod   = (float*)(ws + OFF_DEMOD);
    float*  rn      = (float*)(ws + OFF_RN);
    float*  acc     = (float*)(ws + OFF_ACC);
    float*  Wsq     = (float*)(ws + OFF_WSQ);
    __bf16* WT      = (__bf16*)(ws + OFF_WT);
    __bf16* Bb      = (__bf16*)(ws + OFF_BB);

    k_style<<<1, 512, 0, stream>>>(style, mb_w, mb_b, mod_w, mod_b, cluster, tmpr,
                                   cl_ws, stscale, stvec, acc);
    k_wsq_wt<<<1024, 256, 0, stream>>>(W, Wsq, WT);
    k_rn<<<2, 256, 0, stream>>>(Wsq, rn);
    k_demod<<<16, 256, 0, stream>>>(Wsq, stvec, demod);
    k_bbuf<<<dim3(1024, 9, 8), 512, 0, stream>>>(x, stscale, Bb);
    k_loss<<<1024, 256, 0, stream>>>(co, cl_ws, gate, acc);
    k_gram<<<dim3(32, 32), 256, 0, stream>>>(W, rn, co_out);
    k_conv<<<dim3(4, 8, 32), 256, 0, stream>>>(WT, Bb, demod, noise, flr_bias,
                                               noise_w, out_main);
    k_final<<<1, 1, 0, stream>>>(acc, loss_out);
}
